// ChannelProcessing_18631568130645
// MI455X (gfx1250) — matrix operand order repaired and checked by reading, not compiled, not run on hardware
//
#include <hip/hip_runtime.h>
#include <math.h>

#define B_ 32
#define N_ 784
#define C_ 384
#define HEADS_ 8
#define D_ 48
#define HID_ 1536
#define HW_ 28
#define M_ (B_ * N_)          // 25088 flattened rows, divisible by 256

typedef __bf16 bf16;
typedef __attribute__((ext_vector_type(16))) __bf16 v16bf;
typedef __attribute__((ext_vector_type(8)))  float  v8f;

#define WMMA_BF16(A, Bf, Cacc) \
    __builtin_amdgcn_wmma_f32_16x16x32_bf16(false, (A), false, (Bf), (short)0, (Cacc), false, false)

// ---------------------------------------------------------------------------
// Pack weight [K, Nw] fp32 (row-major, [in,out]) into WMMA B-fragment tiles:
// per (nt,kt) tile of 32x16, lane L holds column n=L&15,
// k = kt*32 + (i&7) + (L>>4)*8 + (i>>3)*16 for i in 0..15 (matches A packing).
// ---------------------------------------------------------------------------
__global__ void pack_w_kernel(const float* __restrict__ src, bf16* __restrict__ dst,
                              int K, int Nw) {
    int idx = blockIdx.x * blockDim.x + threadIdx.x;
    int total = K * Nw;
    if (idx >= total) return;
    int KT = K >> 5;
    int tile = idx >> 9;
    int L = (idx >> 4) & 31;
    int i = idx & 15;
    int nt = tile / KT;
    int kt = tile % KT;
    int k = (kt << 5) + ((L >> 4) << 3) + (i & 7) + ((i >> 3) << 4);
    int n = (nt << 4) + (L & 15);
    dst[idx] = (bf16)src[(size_t)k * Nw + n];
}

// ---------------------------------------------------------------------------
// Pack activation [M_, 384] fp32 into WMMA A-fragment tiles (16x32 bf16):
// lane L (hi=L>>4, r=L&15) holds row m=mt*16+r, elem i -> kk=(i&7)+8*hi+16*(i>>3)
// ---------------------------------------------------------------------------
__global__ void pack_x_kernel(const float* __restrict__ x, bf16* __restrict__ dst) {
    int idx = blockIdx.x * blockDim.x + threadIdx.x;
    const int total = M_ * C_;
    if (idx >= total) return;
    const int KT = C_ / 32;   // 12
    int tile = idx >> 9;
    int L = (idx >> 4) & 31;
    int i = idx & 15;
    int mt = tile / KT;
    int kt = tile % KT;
    int hi = L >> 4;
    int r  = L & 15;
    int kk = (i & 7) + (hi << 3) + ((i >> 3) << 4);
    int m = (mt << 4) + r;
    int k = (kt << 5) + kk;
    dst[idx] = (bf16)x[(size_t)m * C_ + k];
}

// ---------------------------------------------------------------------------
// bf16 WMMA GEMM: C[M_, ldc] = Apack[M_,K] * Bpack[K,ldc] (+bias)
// 8 waves / block; each wave -> 32(M) x 64(N) tile via 8 accumulators,
// register double-buffered across the K loop (6 loads per 8 WMMAs).
// ---------------------------------------------------------------------------
__global__ __launch_bounds__(256)
void gemm_bf16_wmma(const bf16* __restrict__ Ap, const bf16* __restrict__ Bp,
                    void* __restrict__ Cout, const float* __restrict__ bias,
                    int KT, int ldc, int storeBf16) {
    const int lane = threadIdx.x & 31;
    const int wave = threadIdx.x >> 5;
    const int mt0  = blockIdx.y * 16 + wave * 2;   // two 16-row tiles per wave
    const int nt0  = blockIdx.x << 2;

    const bf16* a0 = Ap + (size_t)mt0 * KT * 512 + lane * 16;
    const bf16* a1 = a0 + (size_t)KT * 512;
    const bf16* b0 = Bp + (size_t)nt0 * KT * 512 + lane * 16;
    const bf16* b1 = b0 + (size_t)KT * 512;
    const bf16* b2 = b1 + (size_t)KT * 512;
    const bf16* b3 = b2 + (size_t)KT * 512;

    const v8f zero = {0.f, 0.f, 0.f, 0.f, 0.f, 0.f, 0.f, 0.f};
    v8f acc[8] = {zero, zero, zero, zero, zero, zero, zero, zero};

    // preload kt = 0 fragments
    v16bf A0 = *(const v16bf*)(a0);
    v16bf A1 = *(const v16bf*)(a1);
    v16bf F0 = *(const v16bf*)(b0);
    v16bf F1 = *(const v16bf*)(b1);
    v16bf F2 = *(const v16bf*)(b2);
    v16bf F3 = *(const v16bf*)(b3);

    for (int kt = 0; kt < KT - 1; ++kt) {
        const size_t o = (size_t)(kt + 1) * 512;
        // prefetch A stream two tiles ahead (B panels are L2-resident)
        const size_t po = (size_t)((kt + 2 < KT) ? (kt + 2) : (KT - 1)) * 512;
        __builtin_prefetch(a0 + po, 0, 1);
        __builtin_prefetch(a1 + po, 0, 1);
        // issue next-iteration fragment loads before consuming current ones
        v16bf A0n = *(const v16bf*)(a0 + o);
        v16bf A1n = *(const v16bf*)(a1 + o);
        v16bf F0n = *(const v16bf*)(b0 + o);
        v16bf F1n = *(const v16bf*)(b1 + o);
        v16bf F2n = *(const v16bf*)(b2 + o);
        v16bf F3n = *(const v16bf*)(b3 + o);

        acc[0] = WMMA_BF16(A0, F0, acc[0]);
        acc[4] = WMMA_BF16(A1, F0, acc[4]);
        acc[1] = WMMA_BF16(A0, F1, acc[1]);
        acc[5] = WMMA_BF16(A1, F1, acc[5]);
        acc[2] = WMMA_BF16(A0, F2, acc[2]);
        acc[6] = WMMA_BF16(A1, F2, acc[6]);
        acc[3] = WMMA_BF16(A0, F3, acc[3]);
        acc[7] = WMMA_BF16(A1, F3, acc[7]);

        A0 = A0n; A1 = A1n; F0 = F0n; F1 = F1n; F2 = F2n; F3 = F3n;
    }
    // final K tile
    acc[0] = WMMA_BF16(A0, F0, acc[0]);
    acc[4] = WMMA_BF16(A1, F0, acc[4]);
    acc[1] = WMMA_BF16(A0, F1, acc[1]);
    acc[5] = WMMA_BF16(A1, F1, acc[5]);
    acc[2] = WMMA_BF16(A0, F2, acc[2]);
    acc[6] = WMMA_BF16(A1, F2, acc[6]);
    acc[3] = WMMA_BF16(A0, F3, acc[3]);
    acc[7] = WMMA_BF16(A1, F3, acc[7]);

    const int ncol0 = (nt0 << 4) + (lane & 15);
    float* Cf = (float*)Cout;
    bf16*  Cb = (bf16*)Cout;
#pragma unroll
    for (int mi = 0; mi < 2; ++mi) {
        const int mrow = ((mt0 + mi) << 4) + ((lane >> 4) << 3);
#pragma unroll
        for (int j = 0; j < 4; ++j) {
            int ncol = ncol0 + (j << 4);
            float bb = bias ? bias[ncol] : 0.0f;
#pragma unroll
            for (int r = 0; r < 8; ++r) {
                float v = acc[mi * 4 + j][r] + bb;
                size_t off = (size_t)(mrow + r) * ldc + ncol;
                if (storeBf16) Cb[off] = (bf16)v;
                else           Cf[off] = v;
            }
        }
    }
}

// ---------------------------------------------------------------------------
// Column stats over N for x: per (b,c) -> max_n, sum_n exp(x - max)
// Block handles one (b, group-of-32-columns). threads = 32 cols x 8 rows.
// ---------------------------------------------------------------------------
__global__ __launch_bounds__(256)
void xcolstats_kernel(const float* __restrict__ x,
                      float* __restrict__ colmax, float* __restrict__ colsum) {
    __shared__ float red[256];
    int b  = blockIdx.x / (C_ / 32);
    int cg = blockIdx.x % (C_ / 32);
    int cl = threadIdx.x & 31;
    int nr = threadIdx.x >> 5;
    int c  = (cg << 5) + cl;
    const float* col = x + (size_t)b * N_ * C_ + c;

    float mx = -3.4e38f;
    for (int n = nr; n < N_; n += 8) mx = fmaxf(mx, col[(size_t)n * C_]);
    red[threadIdx.x] = mx; __syncthreads();
    for (int s = 128; s >= 32; s >>= 1) {
        if (threadIdx.x < s) red[threadIdx.x] = fmaxf(red[threadIdx.x], red[threadIdx.x + s]);
        __syncthreads();
    }
    mx = red[cl]; __syncthreads();

    float sum = 0.f;
    for (int n = nr; n < N_; n += 8) sum += expf(col[(size_t)n * C_] - mx);
    red[threadIdx.x] = sum; __syncthreads();
    for (int s = 128; s >= 32; s >>= 1) {
        if (threadIdx.x < s) red[threadIdx.x] += red[threadIdx.x + s];
        __syncthreads();
    }
    if (nr == 0) {
        colmax[(size_t)b * C_ + c] = mx;
        colsum[(size_t)b * C_ + c] = red[cl];
    }
}

// k[b,h,n] = mean_dd softmax_over_n(x)[b,n,h*48+dd]
__global__ void kvec_kernel(const float* __restrict__ x,
                            const float* __restrict__ colmax,
                            const float* __restrict__ colsum,
                            float* __restrict__ kvec) {
    int idx = blockIdx.x * blockDim.x + threadIdx.x;
    const int total = B_ * HEADS_ * N_;
    if (idx >= total) return;
    int n = idx % N_;
    int h = (idx / N_) % HEADS_;
    int b = idx / (N_ * HEADS_);
    const float* xr = x + ((size_t)b * N_ + n) * C_ + h * D_;
    const float* cm = colmax + (size_t)b * C_ + h * D_;
    const float* cs = colsum + (size_t)b * C_ + h * D_;
    float s = 0.f;
    for (int dd = 0; dd < D_; ++dd)
        s += expf(xr[dd] - cm[dd]) / cs[dd];
    kvec[idx] = s * (1.0f / (float)D_);
}

// attn[b,c] = sigmoid( sum_n softmaxN(q)[b,n,c] * k[b,h,n] ) * temp[h]
__global__ __launch_bounds__(256)
void attn_kernel(const float* __restrict__ q, const float* __restrict__ kvec,
                 const float* __restrict__ temp, float* __restrict__ attn) {
    __shared__ float red[256];
    __shared__ float red2[256];
    int b  = blockIdx.x / (C_ / 32);
    int cg = blockIdx.x % (C_ / 32);
    int cl = threadIdx.x & 31;
    int nr = threadIdx.x >> 5;
    int c  = (cg << 5) + cl;
    const float* col = q + (size_t)b * N_ * C_ + c;

    float mx = -3.4e38f;
    for (int n = nr; n < N_; n += 8) mx = fmaxf(mx, col[(size_t)n * C_]);
    red[threadIdx.x] = mx; __syncthreads();
    for (int s = 128; s >= 32; s >>= 1) {
        if (threadIdx.x < s) red[threadIdx.x] = fmaxf(red[threadIdx.x], red[threadIdx.x + s]);
        __syncthreads();
    }
    mx = red[cl]; __syncthreads();

    int h = c / D_;
    const float* kv = kvec + ((size_t)b * HEADS_ + h) * N_;
    float es = 0.f, ek = 0.f;
    for (int n = nr; n < N_; n += 8) {
        float e = expf(col[(size_t)n * C_] - mx);
        es += e;
        ek += e * kv[n];
    }
    red[threadIdx.x] = es; red2[threadIdx.x] = ek; __syncthreads();
    for (int s = 128; s >= 32; s >>= 1) {
        if (threadIdx.x < s) {
            red[threadIdx.x]  += red[threadIdx.x + s];
            red2[threadIdx.x] += red2[threadIdx.x + s];
        }
        __syncthreads();
    }
    if (nr == 0) {
        float v  = red2[cl] / red[cl];
        float sg = 1.0f / (1.0f + expf(-v));
        attn[(size_t)b * C_ + c] = sg * temp[h];
    }
}

// ---------------------------------------------------------------------------
// Depthwise 3x3 conv #1 + bias + exact GELU + BN   (u bf16 -> t bf16)
// ---------------------------------------------------------------------------
__global__ void dwconv1_gelu_bn(const bf16* __restrict__ u, bf16* __restrict__ t,
                                const float* __restrict__ w, const float* __restrict__ bias,
                                const float* __restrict__ bnw, const float* __restrict__ bnb,
                                const float* __restrict__ bnm, const float* __restrict__ bnv) {
    int idx = blockIdx.x * blockDim.x + threadIdx.x;
    const int total = B_ * N_ * HID_;
    if (idx >= total) return;
    int hid = idx % HID_;
    int rem = idx / HID_;
    int n = rem % N_;
    int b = rem / N_;
    int y = n / HW_, x = n % HW_;
    float s = 0.f;
#pragma unroll
    for (int dy = -1; dy <= 1; ++dy) {
        int yy = y + dy;
        if (yy < 0 || yy >= HW_) continue;
#pragma unroll
        for (int dx = -1; dx <= 1; ++dx) {
            int xx = x + dx;
            if (xx < 0 || xx >= HW_) continue;
            float uv = (float)u[((size_t)b * N_ + yy * HW_ + xx) * HID_ + hid];
            s += uv * w[hid * 9 + (dy + 1) * 3 + (dx + 1)];
        }
    }
    s += bias[hid];
    s = 0.5f * s * (1.0f + erff(s * 0.70710678118654752f));                 // exact GELU
    s = bnw[hid] * (s - bnm[hid]) * rsqrtf(bnv[hid] + 1e-5f) + bnb[hid];    // BN
    t[idx] = (bf16)s;
}

// ---------------------------------------------------------------------------
// Depthwise 3x3 conv #2 + bias, z = gamma*dw + u; write z directly into the
// packed WMMA A-fragment layout for GEMM3 (K = HID, KT = 48).
// ---------------------------------------------------------------------------
__global__ void dwconv2_res_pack(const bf16* __restrict__ t, const bf16* __restrict__ u,
                                 bf16* __restrict__ zp,
                                 const float* __restrict__ w, const float* __restrict__ bias,
                                 const float* __restrict__ gamma) {
    int idx = blockIdx.x * blockDim.x + threadIdx.x;
    const int total = B_ * N_ * HID_;
    if (idx >= total) return;
    int hid = idx % HID_;
    int rem = idx / HID_;
    int n = rem % N_;
    int b = rem / N_;
    int y = n / HW_, x = n % HW_;
    float s = 0.f;
#pragma unroll
    for (int dy = -1; dy <= 1; ++dy) {
        int yy = y + dy;
        if (yy < 0 || yy >= HW_) continue;
#pragma unroll
        for (int dx = -1; dx <= 1; ++dx) {
            int xx = x + dx;
            if (xx < 0 || xx >= HW_) continue;
            float tv = (float)t[((size_t)b * N_ + yy * HW_ + xx) * HID_ + hid];
            s += tv * w[hid * 9 + (dy + 1) * 3 + (dx + 1)];
        }
    }
    s += bias[hid];
    float z = gamma[hid] * s + (float)u[idx];

    // packed A index for (m = b*N_+n, k = hid), KT = HID_/32 = 48
    int m  = b * N_ + n;
    int mt = m >> 4;
    int r  = m & 15;
    int kt = hid >> 5;
    int kk = hid & 31;
    int L  = r + (((kk >> 3) & 1) << 4);
    int i  = (kk & 7) + ((kk >> 4) << 3);
    zp[(((size_t)mt * (HID_ / 32) + kt) * 32 + L) * 16 + i] = (bf16)z;
}

// ---------------------------------------------------------------------------
// LayerNorm over C + attn scaling; writes both outputs.
// One 128-thread block per row (b,n); 3 channels per thread.
// ---------------------------------------------------------------------------
__global__ __launch_bounds__(128)
void ln_scale_out(const float* __restrict__ u2, const float* __restrict__ attn,
                  const float* __restrict__ lnw, const float* __restrict__ lnb,
                  float* __restrict__ out1, float* __restrict__ out2) {
    __shared__ float ss[128];
    __shared__ float sq[128];
    int row = blockIdx.x;            // b*N_ + n
    int b = row / N_, n = row % N_;
    int tid = threadIdx.x;
    const float* ur = u2 + (size_t)row * C_;
    float v0 = ur[tid], v1 = ur[tid + 128], v2 = ur[tid + 256];
    ss[tid] = v0 + v1 + v2;
    sq[tid] = v0 * v0 + v1 * v1 + v2 * v2;
    __syncthreads();
    for (int s = 64; s > 0; s >>= 1) {
        if (tid < s) { ss[tid] += ss[tid + s]; sq[tid] += sq[tid + s]; }
        __syncthreads();
    }
    float mean = ss[0] * (1.0f / (float)C_);
    float var  = sq[0] * (1.0f / (float)C_) - mean * mean;
    float inv  = rsqrtf(var + 1e-5f);
    const float* arow = attn + (size_t)b * C_;
    float* o1 = out1 + (size_t)row * C_;
    float vals[3] = {v0, v1, v2};
#pragma unroll
    for (int e = 0; e < 3; ++e) {
        int c = tid + e * 128;
        float val = (vals[e] - mean) * inv * lnw[c] + lnb[c];
        float o = arow[c] * val;
        o1[c] = o;
        int h = c / D_, dd = c % D_;
        out2[(((size_t)b * HEADS_ + h) * N_ + n) * D_ + dd] = o;
    }
}

// ---------------------------------------------------------------------------
extern "C" void kernel_launch(void* const* d_in, const int* in_sizes, int n_in,
                              void* d_out, int out_size, void* d_ws, size_t ws_size,
                              hipStream_t stream) {
    const float* x      = (const float*)d_in[0];
    const float* Wq     = (const float*)d_in[1];
    const float* temp   = (const float*)d_in[2];
    const float* fc1_w  = (const float*)d_in[3];
    const float* fc1_b  = (const float*)d_in[4];
    const float* gamma  = (const float*)d_in[5];
    const float* conv1w = (const float*)d_in[6];
    const float* conv1b = (const float*)d_in[7];
    const float* bn_w   = (const float*)d_in[8];
    const float* bn_b   = (const float*)d_in[9];
    const float* bn_m   = (const float*)d_in[10];
    const float* bn_v   = (const float*)d_in[11];
    const float* conv2w = (const float*)d_in[12];
    const float* conv2b = (const float*)d_in[13];
    const float* fc2_w  = (const float*)d_in[14];
    const float* fc2_b  = (const float*)d_in[15];
    const float* ln_w   = (const float*)d_in[16];
    const float* ln_b   = (const float*)d_in[17];

    float* out1 = (float*)d_out;
    float* out2 = out1 + (size_t)B_ * N_ * C_;

    // workspace carve-out (256B aligned)
    char* base = (char*)d_ws;
    size_t off = 0;
    auto carve = [&](size_t bytes) -> void* {
        off = (off + 255) & ~(size_t)255;
        void* p = base + off;
        off += bytes;
        return p;
    };
    bf16*  wqp   = (bf16*) carve((size_t)C_ * C_ * 2);
    bf16*  fc1p  = (bf16*) carve((size_t)C_ * HID_ * 2);
    bf16*  fc2p  = (bf16*) carve((size_t)HID_ * C_ * 2);
    bf16*  xap   = (bf16*) carve((size_t)M_ * C_ * 2);
    float* cmax  = (float*)carve((size_t)B_ * C_ * 4);
    float* csum  = (float*)carve((size_t)B_ * C_ * 4);
    float* kvec  = (float*)carve((size_t)B_ * HEADS_ * N_ * 4);
    float* qbuf  = (float*)carve((size_t)M_ * C_ * 4);      // reused as GEMM3 output
    float* attnb = (float*)carve((size_t)B_ * C_ * 4);
    bf16*  ubuf  = (bf16*) carve((size_t)M_ * HID_ * 2);
    bf16*  tbuf  = (bf16*) carve((size_t)M_ * HID_ * 2);
    bf16*  zp    = (bf16*) carve((size_t)M_ * HID_ * 2);
    (void)ws_size;

    const int TPB = 256;

    // 1) pack weights and activations into WMMA fragment layouts
    pack_w_kernel<<<(C_ * C_ + TPB - 1) / TPB, TPB, 0, stream>>>(Wq, wqp, C_, C_);
    pack_w_kernel<<<(C_ * HID_ + TPB - 1) / TPB, TPB, 0, stream>>>(fc1_w, fc1p, C_, HID_);
    pack_w_kernel<<<(HID_ * C_ + TPB - 1) / TPB, TPB, 0, stream>>>(fc2_w, fc2p, HID_, C_);
    pack_x_kernel<<<(M_ * C_ + TPB - 1) / TPB, TPB, 0, stream>>>(x, xap);

    // 2) key path: column softmax stats + k vector
    xcolstats_kernel<<<B_ * (C_ / 32), TPB, 0, stream>>>(x, cmax, csum);
    kvec_kernel<<<(B_ * HEADS_ * N_ + TPB - 1) / TPB, TPB, 0, stream>>>(x, cmax, csum, kvec);

    // 3) GEMM1: q = x @ Wq (no bias), fp32 out
    gemm_bf16_wmma<<<dim3(C_ / 64, M_ / 256), TPB, 0, stream>>>(
        xap, wqp, (void*)qbuf, nullptr, C_ / 32, C_, 0);

    // 4) attn = sigmoid(softmaxN(q) . k) * temperature
    attn_kernel<<<B_ * (C_ / 32), TPB, 0, stream>>>(qbuf, kvec, temp, attnb);

    // 5) GEMM2: u = x @ fc1 + b, bf16 out
    gemm_bf16_wmma<<<dim3(HID_ / 64, M_ / 256), TPB, 0, stream>>>(
        xap, fc1p, (void*)ubuf, fc1_b, C_ / 32, HID_, 1);

    // 6) dwconv1 + GELU + BN  ->  t
    dwconv1_gelu_bn<<<(B_ * N_ * HID_ + TPB - 1) / TPB, TPB, 0, stream>>>(
        ubuf, tbuf, conv1w, conv1b, bn_w, bn_b, bn_m, bn_v);

    // 7) dwconv2 + residual, written directly in packed-A layout -> zp
    dwconv2_res_pack<<<(B_ * N_ * HID_ + TPB - 1) / TPB, TPB, 0, stream>>>(
        tbuf, ubuf, zp, conv2w, conv2b, gamma);

    // 8) GEMM3: u2 = z @ fc2 + b, fp32 out (reuse q buffer)
    gemm_bf16_wmma<<<dim3(C_ / 64, M_ / 256), TPB, 0, stream>>>(
        zp, fc2p, (void*)qbuf, fc2_b, HID_ / 32, C_, 0);

    // 9) LayerNorm + attn scaling -> both outputs
    ln_scale_out<<<M_, 128, 0, stream>>>(qbuf, attnb, ln_w, ln_b, out1, out2);
    (void)in_sizes; (void)n_in; (void)out_size;
}
